// SimpleGateLoopLayer_11287174054357
// MI455X (gfx1250) — matrix operand order repaired and checked by reading, not compiled, not run on hardware
//
#include <hip/hip_runtime.h>
#include <math.h>
#include <stdint.h>

typedef __bf16 bf16_t;
typedef __attribute__((ext_vector_type(16))) __bf16 v16bf;
typedef __attribute__((ext_vector_type(8)))  __bf16 v8bf;
typedef __attribute__((ext_vector_type(4)))  __bf16 v4bf;
typedef __attribute__((ext_vector_type(8)))  float  v8f;

#define DIM     1024
#define BATCH   4
#define SEQ     4096
#define M_TOTAL (BATCH * SEQ)   // 16384 rows
#define E_TOTAL (3 * DIM)       // 3072 cols (q | kv | a)

// GEMM tiling: block 128x256 (8 waves, 2x4), wave 64x64 (4x4 WMMA frags), BK=64
#define BM 128
#define BN 256
#define BK 64
#define LDS_PITCH 72            // bf16 elements; 144 B pitch (16B-aligned, bank-spread)

// ---------------------------------------------------------------------------
// CDNA5 async-copy helpers (gfx1250): global -> LDS without touching VGPRs,
// tracked by ASYNCcnt.
// ---------------------------------------------------------------------------
__device__ __forceinline__ void async_copy_b128(uint32_t lds_byte_addr, const void* gaddr) {
    asm volatile("global_load_async_to_lds_b128 %0, %1, off"
                 :: "v"(lds_byte_addr), "v"((uint64_t)(size_t)gaddr)
                 : "memory");
}
__device__ __forceinline__ void wait_async0() {
    asm volatile("s_wait_asynccnt 0x0" ::: "memory");
}
__device__ __forceinline__ uint32_t lds_addr_of(const void* p) {
    // low 32 bits of a generic pointer into LDS == LDS byte address
    return (uint32_t)(size_t)p;
}

// ---------------------------------------------------------------------------
// Kernel 1: RMSNorm + fp32 -> bf16 down-convert.  One block per row.
// ---------------------------------------------------------------------------
__global__ __launch_bounds__(256)
void k_rmsnorm_bf16(const float* __restrict__ x,
                    const float* __restrict__ gamma,
                    bf16_t* __restrict__ h) {
    const int row = blockIdx.x;
    const int tid = threadIdx.x;

    const float4* xr = (const float4*)(x + (size_t)row * DIM);
    float4 v = xr[tid];
    float ss = v.x * v.x + v.y * v.y + v.z * v.z + v.w * v.w;

    __shared__ float red[256];
    red[tid] = ss;
    __syncthreads();
    for (int s = 128; s > 0; s >>= 1) {
        if (tid < s) red[tid] += red[tid + s];
        __syncthreads();
    }
    const float norm  = sqrtf(red[0]);
    const float scale = 32.0f / fmaxf(norm, 1e-12f);   // sqrt(1024) = 32

    const float4 g = ((const float4*)gamma)[tid];
    v4bf o;
    o[0] = (bf16_t)(v.x * scale * g.x);
    o[1] = (bf16_t)(v.y * scale * g.y);
    o[2] = (bf16_t)(v.z * scale * g.z);
    o[3] = (bf16_t)(v.w * scale * g.w);
    *(v4bf*)(h + (size_t)row * DIM + (size_t)tid * 4) = o;
}

// ---------------------------------------------------------------------------
// Kernel 2: fp32 -> bf16 convert of the weight matrix [3072, 1024].
// ---------------------------------------------------------------------------
__global__ __launch_bounds__(256)
void k_convert_w(const float* __restrict__ w, bf16_t* __restrict__ wb) {
    const size_t i = (size_t)blockIdx.x * blockDim.x + threadIdx.x; // float4 units
    const float4 v = ((const float4*)w)[i];
    v4bf o;
    o[0] = (bf16_t)v.x;
    o[1] = (bf16_t)v.y;
    o[2] = (bf16_t)v.z;
    o[3] = (bf16_t)v.w;
    ((v4bf*)wb)[i] = o;
}

// ---------------------------------------------------------------------------
// Kernel 3: GEMM  C[16384, 3072] = A[16384,1024](bf16) * W[3072,1024](bf16)^T
// Double-buffered LDS staging via global_load_async_to_lds_b128 (ASYNCcnt),
// fp32 accumulate via V_WMMA_F32_16X16X32_BF16.
// Block = 8 waves, tile 128x256; wave tile 64x64 (4x4 frags, 16 WMMA/k-step).
// ---------------------------------------------------------------------------
__global__ __launch_bounds__(256)
void k_gemm_wmma(const bf16_t* __restrict__ A,
                 const bf16_t* __restrict__ W,
                 float* __restrict__ C) {
    __shared__ bf16_t sA[2][BM * LDS_PITCH];   // 2 * 128*72*2 = 36 KB
    __shared__ bf16_t sB[2][BN * LDS_PITCH];   // 2 * 256*72*2 = 72 KB

    const int t     = threadIdx.x;
    const int lane  = t & 31;
    const int wave  = t >> 5;
    const int half  = lane >> 4;     // 0 | 1
    const int r     = lane & 15;     // 0..15
    const int waveM = wave & 1;      // 2 waves along M
    const int waveN = wave >> 1;     // 4 waves along N

    const int mBlock = blockIdx.x * BM;
    const int nBlock = blockIdx.y * BN;

    // Cooperative load mapping: 8 threads per 128-B row (BK=64 bf16), 16 B each.
    const int lrow = t >> 3;          // 0..31
    const int lcol = (t & 7) * 8;     // bf16 element offset within the k-chunk

    auto issue_stage = [&](int buf, int k0) {
        // A tile: 128 rows -> 4 passes of 32 rows
#pragma unroll
        for (int p = 0; p < 4; ++p) {
            const int row = lrow + p * 32;
            async_copy_b128(lds_addr_of(&sA[buf][row * LDS_PITCH + lcol]),
                            A + (size_t)(mBlock + row) * DIM + k0 + lcol);
        }
        // B tile: 256 rows -> 8 passes of 32 rows
#pragma unroll
        for (int p = 0; p < 8; ++p) {
            const int row = lrow + p * 32;
            async_copy_b128(lds_addr_of(&sB[buf][row * LDS_PITCH + lcol]),
                            W + (size_t)(nBlock + row) * DIM + k0 + lcol);
        }
    };

    v8f acc[4][4] = {};
    union f16x16 { v16bf v; v8bf h[2]; };

    issue_stage(0, 0);

    const int NSTAGE = DIM / BK;      // 16
    for (int s = 0; s < NSTAGE; ++s) {
        const int buf = s & 1;
        wait_async0();                // this wave's loads for stage s complete
        __syncthreads();              // all waves' loads complete; stage s-1 compute done
        if (s + 1 < NSTAGE) issue_stage(buf ^ 1, (s + 1) * BK);  // overlap with compute

        const bf16_t* aBase = &sA[buf][(waveM * 64 + r) * LDS_PITCH];
        const bf16_t* bBase = &sB[buf][(waveN * 64 + r) * LDS_PITCH];

#pragma unroll
        for (int kk = 0; kk < BK; kk += 32) {
            f16x16 fa[4];
            f16x16 fb[4];
#pragma unroll
            for (int i = 0; i < 4; ++i) {
                const bf16_t* ap = aBase + i * 16 * LDS_PITCH + kk;
                fa[i].h[0] = *(const v8bf*)(ap + half * 8);
                fa[i].h[1] = *(const v8bf*)(ap + 16 + half * 8);
            }
#pragma unroll
            for (int j = 0; j < 4; ++j) {
                fb[j].h[0] = *(const v8bf*)(bBase + j * 16 * LDS_PITCH + kk + half * 8);
                fb[j].h[1] = *(const v8bf*)(bBase + j * 16 * LDS_PITCH + kk + 16 + half * 8);
            }
#pragma unroll
            for (int i = 0; i < 4; ++i) {
#pragma unroll
                for (int j = 0; j < 4; ++j) {
                    acc[i][j] = __builtin_amdgcn_wmma_f32_16x16x32_bf16(
                        false, fa[i].v, false, fb[j].v, (short)0, acc[i][j], false, false);
                }
            }
        }
        // single barrier per stage (at loop top): a wave reaches it only after
        // finishing compute on the buffer the next issue will overwrite
    }

    // C/D layout: c[v] -> row (half*8 + v), col r
    const int mWave = mBlock + waveM * 64;
    const int nWave = nBlock + waveN * 64;
#pragma unroll
    for (int i = 0; i < 4; ++i) {
        const int rowB = mWave + i * 16 + half * 8;
#pragma unroll
        for (int j = 0; j < 4; ++j) {
            const int col = nWave + j * 16 + r;
#pragma unroll
            for (int v = 0; v < 8; ++v) {
                C[(size_t)(rowB + v) * E_TOTAL + col] = acc[i][j][v];
            }
        }
    }
}

// ---------------------------------------------------------------------------
// Kernel 4: sigmoid gate + linear recurrence y[n] = a[n]*y[n-1] + kv[n],
// out = q * y.  One thread per (batch, channel): 4096 independent chains.
// ---------------------------------------------------------------------------
__global__ __launch_bounds__(256)
void k_gate_scan(const float* __restrict__ qkva, float* __restrict__ out) {
    const int t = blockIdx.x * 256 + threadIdx.x;   // 0..4095
    const int b = t >> 10;                          // batch
    const int d = t & (DIM - 1);                    // channel

    const float* base = qkva + (size_t)b * SEQ * E_TOTAL;
    float* ob = out + (size_t)b * SEQ * DIM + d;

    float y = 0.0f;
    for (int n = 0; n < SEQ; ++n) {
        const float* p = base + (size_t)n * E_TOTAL;
        if (n + 8 < SEQ) {  // 12 KB row stride: prefetch ahead to hide latency
            const float* pf = base + (size_t)(n + 8) * E_TOTAL;
            __builtin_prefetch(pf + d,           0, 1);
            __builtin_prefetch(pf + DIM + d,     0, 1);
            __builtin_prefetch(pf + 2 * DIM + d, 0, 1);
        }
        const float q  = p[d];
        const float kv = p[DIM + d];
        const float ag = p[2 * DIM + d];
        const float a  = 1.0f / (1.0f + __expf(-ag));
        y = fmaf(a, y, kv);
        ob[(size_t)n * DIM] = q * y;
    }
}

// ---------------------------------------------------------------------------
// Launch
// ---------------------------------------------------------------------------
extern "C" void kernel_launch(void* const* d_in, const int* in_sizes, int n_in,
                              void* d_out, int out_size, void* d_ws, size_t ws_size,
                              hipStream_t stream) {
    (void)in_sizes; (void)n_in; (void)out_size; (void)ws_size;

    const float* x     = (const float*)d_in[0];
    const float* w     = (const float*)d_in[1];
    const float* gamma = (const float*)d_in[2];
    float* out = (float*)d_out;

    char* ws = (char*)d_ws;
    const size_t H_BYTES = (size_t)M_TOTAL * DIM * sizeof(bf16_t);   // 32 MB
    const size_t W_BYTES = (size_t)E_TOTAL * DIM * sizeof(bf16_t);   //  6 MB
    bf16_t* h_bf = (bf16_t*)ws;
    bf16_t* w_bf = (bf16_t*)(ws + H_BYTES);
    float*  qkva = (float*)(ws + H_BYTES + W_BYTES);                 // 192 MB

    k_rmsnorm_bf16<<<M_TOTAL, 256, 0, stream>>>(x, gamma, h_bf);
    k_convert_w<<<(E_TOTAL * DIM / 4) / 256, 256, 0, stream>>>(w, w_bf);

    dim3 grid(M_TOTAL / BM, E_TOTAL / BN);   // 128 x 12 blocks
    k_gemm_wmma<<<grid, 256, 0, stream>>>(h_bf, w_bf, qkva);

    k_gate_scan<<<(BATCH * DIM) / 256, 256, 0, stream>>>(qkva, out);
}
